// q_RNN_36953898615493
// MI455X (gfx1250) — matrix operand order repaired and checked by reading, not compiled, not run on hardware
//
#include <hip/hip_runtime.h>
#include <hip/hip_bf16.h>
#include <math.h>

// ---------------------------------------------------------------------------
// MI455X (gfx1250) implementation plan:
//   K_cvt   : f32 -> bf16 with K-padding (data, W_lat, W_ih, W_hh)
//   K_gemm1 : X[4096x400](bf16) = data_bf @ W_lat_bf^T + b_lat      (WMMA bf16)
//   K_gemm2 : gates_x[4096x1600](f32) = X @ W_ih^T + b_ih + b_hh    (WMMA bf16)
//             GEMMs register-block 4 M-tiles/wave: B-frag reused by 4 WMMAs.
//   K_lstm  : 512 sequential steps, single 1024-thread WG; per step
//             g = h @ W_hh^T (WMMA, h kept in LDS as padded 16x416 bf16 tile,
//             A-frag reused across 4 N-tiles), gate nonlinearity + c/h update
//             in LDS; h[batch 0] -> H0[t]
//   K_proj  : a_mu/a_lv = H0 @ Wm_o^T / Wl_o^T (scalar f32, tiny)
//   K_scan  : 50-dim latent scan, 511 steps, 1 block (scalar f32, tiny)
// ---------------------------------------------------------------------------

typedef __bf16 bf16;
typedef __attribute__((ext_vector_type(16))) __bf16 v16bf;
typedef __attribute__((ext_vector_type(8)))  __bf16 v8bf;
typedef __attribute__((ext_vector_type(8)))  float  v8f;

union V16 { v16bf v; v8bf h[2]; };

__device__ __forceinline__ float sigmoidf(float x) { return 1.0f / (1.0f + expf(-x)); }

// ----------------------------- convert / pad -------------------------------
__global__ void k_cvt_pad(const float* __restrict__ src, bf16* __restrict__ dst,
                          int rows, int K, int Kp)
{
    long long idx = (long long)blockIdx.x * blockDim.x + threadIdx.x;
    long long total = (long long)rows * Kp;
    if (idx >= total) return;
    int r = (int)(idx / Kp);
    int k = (int)(idx % Kp);
    dst[idx] = (k < K) ? (bf16)src[(size_t)r * K + k] : (bf16)0.0f;
}

__global__ void k_fill_zero_bf16(bf16* __restrict__ p, long long n)
{
    long long idx = (long long)blockIdx.x * blockDim.x + threadIdx.x;
    if (idx < n) p[idx] = (bf16)0.0f;
}

// ------------------------------- WMMA GEMM ---------------------------------
// C[M x N] = A[M x Kp](bf16) * W[N x Kp](bf16)^T + bias1 + bias2
// One 64x16 output block per wave (4 stacked 16x16 tiles): the B (weight)
// fragment is loaded once per K-step and reused by 4 WMMAs.
template <bool OUT_BF16>
__global__ void k_gemm_wmma(const bf16* __restrict__ A, int lda,
                            const bf16* __restrict__ W, int ldw,
                            const float* __restrict__ bias1,
                            const float* __restrict__ bias2,
                            void* __restrict__ Cout, int ldc,
                            int Mgroups, int Ntiles, int Ksteps)   // Mgroups = Mtiles/4
{
    const int wavesPerBlock = blockDim.x >> 5;
    const int wave = blockIdx.x * wavesPerBlock + (threadIdx.x >> 5);
    const int lane = threadIdx.x & 31;
    if (wave >= Mgroups * Ntiles) return;          // wave-uniform
    const int gm = wave / Ntiles;                  // 64-row group
    const int nt = wave % Ntiles;

    // ISA 7.12.2 layouts (wave32):
    // A 16x32 bf16: lane = 16*half + m ; elems[0..7]=K kb..kb+7, elems[8..15]=K kb+16..kb+23, kb=8*half
    // B 32x16 bf16 (B = W^T): lane = 16*half + n ; elems[0..7]=W[n][k0+kb..kb+7], elems[8..15]=W[n][k0+kb+16..kb+23], kb=8*half
    const int rowA = lane & 15;
    const int kbA  = (lane >> 4) * 8;
    const int kbB  = (lane >> 4) * 8;

    const bf16* abase = A + (size_t)(gm * 64 + rowA) * lda;   // + i*16*lda per tile
    const bf16* wrow  = W + (size_t)(nt * 16 + (lane & 15)) * ldw;

    v8f acc[4] = {{}, {}, {}, {}};
    for (int ks = 0; ks < Ksteps; ++ks) {
        const int k0 = ks * 32;
        V16 b;
        b.h[0] = *(const v8bf*)(wrow + k0 + kbB);
        b.h[1] = *(const v8bf*)(wrow + k0 + kbB + 16);
#pragma unroll
        for (int i = 0; i < 4; ++i) {
            const bf16* arow = abase + (size_t)i * 16 * lda;
            V16 a;
            a.h[0] = *(const v8bf*)(arow + k0 + kbA);
            a.h[1] = *(const v8bf*)(arow + k0 + kbA + 16);
            acc[i] = __builtin_amdgcn_wmma_f32_16x16x32_bf16(false, a.v, false, b.v,
                                                             (short)0, acc[i], false, false);
        }
    }

    // C/D layout: VGPR r -> row r + 8*(lane>=16); col = lane&15
    const int colC   = lane & 15;
    const int rowOff = (lane >> 4) * 8;
    const int gcol   = nt * 16 + colC;
    float bsum = bias1 ? bias1[gcol] : 0.0f;
    if (bias2) bsum += bias2[gcol];
#pragma unroll
    for (int i = 0; i < 4; ++i) {
#pragma unroll
        for (int r = 0; r < 8; ++r) {
            const int grow = gm * 64 + i * 16 + r + rowOff;
            const float v = acc[i][r] + bsum;
            if (OUT_BF16) ((bf16*)Cout)[(size_t)grow * ldc + gcol] = (bf16)v;
            else          ((float*)Cout)[(size_t)grow * ldc + gcol] = v;
        }
    }
}

// ------------------------------- LSTM scan ---------------------------------
// Single workgroup, 1024 threads (32 waves). Per step t:
//   g[8x1600] = h[16x416](LDS,bf16, rows 8..15 zero) @ W_hh^T  via WMMA
//   (25 groups of 4 N-tiles; the h A-fragment is loaded once per K-step and
//    reused by 4 WMMAs to cut LDS traffic on the serial critical path)
//   gate nonlinearity, c/h update in LDS, h(batch 0) -> H0[t]
__global__ void k_lstm_seq(const bf16* __restrict__ Whh,   // [1600 x 416] bf16
                           const float* __restrict__ gx,   // [4096 x 1600] f32, row=(b*512+t)
                           float* __restrict__ H0,         // [512 x 400]
                           int L)
{
    extern __shared__ char smem[];
    float* g_s = (float*)smem;               // 8*1600 f32  = 51200 B
    float* c_s = g_s + 8 * 1600;             // 8*400  f32  = 12800 B
    bf16*  h_s = (bf16*)(c_s + 8 * 400);     // 16*416 bf16 = 13312 B

    const int tid  = threadIdx.x;
    const int lane = tid & 31;
    const int wave = tid >> 5;

    for (int i = tid; i < 8 * 400; i += blockDim.x)  c_s[i] = 0.0f;
    for (int i = tid; i < 16 * 416; i += blockDim.x) h_s[i] = (bf16)0.0f;
    __syncthreads();

    const int rowA   = lane & 15;
    const int kbA    = (lane >> 4) * 8;
    const int kbB    = (lane >> 4) * 8;
    const int colC   = lane & 15;
    const int rowOff = (lane >> 4) * 8;

    for (int t = 0; t < L; ++t) {
        // ---- g = h @ Whh^T : 25 groups of 4 column tiles over 32 waves ----
        for (int g = wave; g < 25; g += 32) {                   // waves 25..31 idle
            const bf16* wbase = Whh + (size_t)(g * 64 + (lane & 15)) * 416;
            v8f acc[4] = {{}, {}, {}, {}};
#pragma unroll
            for (int k0 = 0; k0 < 416; k0 += 32) {
                V16 a;
                a.h[0] = *(const v8bf*)(h_s + rowA * 416 + k0 + kbA);
                a.h[1] = *(const v8bf*)(h_s + rowA * 416 + k0 + kbA + 16);
#pragma unroll
                for (int j = 0; j < 4; ++j) {
                    const bf16* wrow = wbase + (size_t)j * 16 * 416;
                    V16 b;
                    b.h[0] = *(const v8bf*)(wrow + k0 + kbB);
                    b.h[1] = *(const v8bf*)(wrow + k0 + kbB + 16);
                    acc[j] = __builtin_amdgcn_wmma_f32_16x16x32_bf16(false, a.v, false, b.v,
                                                                     (short)0, acc[j], false, false);
                }
            }
#pragma unroll
            for (int j = 0; j < 4; ++j) {
                const int gcol = (g * 4 + j) * 16 + colC;
#pragma unroll
                for (int r = 0; r < 8; ++r) {
                    const int row = r + rowOff;        // lanes>=16 hold rows 8..15 (zero h)
                    if (row < 8)
                        g_s[row * 1600 + gcol] =
                            acc[j][r] + gx[(size_t)(row * 512 + t) * 1600 + gcol];
                }
            }
        }
        __syncthreads();

        // ---- gates -> c,h ----
        for (int idx = tid; idx < 8 * 400; idx += blockDim.x) {
            const int b = idx / 400, j = idx % 400;
            const float gi = g_s[b * 1600 + j];
            const float gf = g_s[b * 1600 + 400 + j];
            const float gg = g_s[b * 1600 + 800 + j];
            const float go = g_s[b * 1600 + 1200 + j];
            const float c  = sigmoidf(gf) * c_s[idx] + sigmoidf(gi) * tanhf(gg);
            c_s[idx] = c;
            const float h = sigmoidf(go) * tanhf(c);
            h_s[b * 416 + j] = (bf16)h;
            if (b == 0) H0[(size_t)t * 400 + j] = h;
        }
        __syncthreads();
    }
}

// ------------------------- latent projections ------------------------------
__global__ void k_proj(const float* __restrict__ H0,
                       const float* __restrict__ Wmu, const float* __restrict__ bmu,
                       const float* __restrict__ Wlv, const float* __restrict__ blv,
                       float* __restrict__ amu, float* __restrict__ alv)
{
    const int idx = blockIdx.x * blockDim.x + threadIdx.x;   // 512*50
    if (idx >= 512 * 50) return;
    const int t = idx / 50, i = idx % 50;
    const float* h  = H0  + (size_t)t * 400;
    const float* wm = Wmu + (size_t)i * 450;   // W_mu[:, :400]
    const float* wl = Wlv + (size_t)i * 450;
    float sm = bmu[i], sl = blv[i];
    for (int j = 0; j < 400; ++j) { sm += h[j] * wm[j]; sl += h[j] * wl[j]; }
    amu[idx] = sm;
    alv[idx] = sl;
}

// ----------------------------- latent scan ---------------------------------
// out layout: [0,25600) z ; [25600,25650) mus[-1] ; [25650,25700) lvs[-1] ; [25700] kl
__global__ void k_scan(const float* __restrict__ amu, const float* __restrict__ alv,
                       const float* __restrict__ eps,
                       const float* __restrict__ Wmu, const float* __restrict__ Wlv,
                       float* __restrict__ out)
{
    __shared__ float zsh[50];
    __shared__ float klsh[64];
    const int i = threadIdx.x;                 // block of 64
    const float DELTA = 0.005f;
    const float log_delta = -5.2983174f;       // ln(0.005)
    float klacc = 0.0f;
    float mu_last = 0.0f, lv_last = 0.0f;

    if (i < 50) {
        const float mu0 = amu[i], lv0 = alv[i];
        const float z0 = mu0 + eps[i] * expf(0.5f * lv0);
        out[i] = z0;
        zsh[i] = z0;
        klacc += -0.5f * (1.0f + lv0 - mu0 * mu0 - expf(lv0));
    }
    __syncthreads();

    for (int t = 1; t < 512; ++t) {
        float z = 0.0f;
        if (i < 50) {
            float dm = 0.0f, dl = 0.0f;
            const float* wm = Wmu + (size_t)i * 450 + 400;  // W_mu[:, 400:]
            const float* wl = Wlv + (size_t)i * 450 + 400;
            for (int j = 0; j < 50; ++j) {
                const float zp = zsh[j];
                dm += zp * wm[j];
                dl += zp * wl[j];
            }
            const float mu = amu[t * 50 + i] + dm;
            const float lv = fminf(alv[t * 50 + i] + dl, 100.0f);
            const float zprev = zsh[i];
            z = mu + eps[t * 50 + i] * expf(0.5f * lv);
            klacc += -0.5f * (1.0f - log_delta + lv -
                              ((mu - zprev) * (mu - zprev) + expf(lv)) / DELTA);
            out[t * 50 + i] = z;
            if (t == 511) { mu_last = mu; lv_last = lv; }
        }
        __syncthreads();
        if (i < 50) zsh[i] = z;
        __syncthreads();
    }

    if (i < 50) { out[25600 + i] = mu_last; out[25650 + i] = lv_last; }
    klsh[i] = klacc;
    __syncthreads();
    if (i == 0) {
        float s = 0.0f;
        for (int j = 0; j < 64; ++j) s += klsh[j];
        out[25700] = s;
    }
}

// ------------------------------- launcher ----------------------------------
extern "C" void kernel_launch(void* const* d_in, const int* in_sizes, int n_in,
                              void* d_out, int out_size, void* d_ws, size_t ws_size,
                              hipStream_t stream)
{
    (void)in_sizes; (void)n_in; (void)out_size; (void)ws_size;
    const float* data  = (const float*)d_in[0];   // (8,512,5000)
    const float* eps   = (const float*)d_in[1];   // (512,50)
    const float* W_lat = (const float*)d_in[2];   // (400,5000)
    const float* b_lat = (const float*)d_in[3];   // (400,)
    const float* W_ih  = (const float*)d_in[4];   // (1600,400)
    const float* W_hh  = (const float*)d_in[5];   // (1600,400)
    const float* b_ih  = (const float*)d_in[6];   // (1600,)
    const float* b_hh  = (const float*)d_in[7];   // (1600,)
    const float* W_mu  = (const float*)d_in[8];   // (50,450)
    const float* b_mu  = (const float*)d_in[9];   // (50,)
    const float* W_lv  = (const float*)d_in[10];  // (50,450)
    const float* b_lv  = (const float*)d_in[11];  // (50,)
    float* out = (float*)d_out;

    const int M   = 4096;            // B*L
    const int D   = 5000, Kp1 = 5024;
    const int Hl  = 400,  Kp2 = 416;
    const int G   = 1600;
    const int L   = 512;

    // workspace carve-up (256B aligned)
    char* ws = (char*)d_ws;
    size_t off = 0;
    auto carve = [&](size_t bytes) -> char* {
        char* p = ws + off;
        off += (bytes + 255) & ~(size_t)255;
        return p;
    };
    bf16*  data_bf = (bf16*)carve((size_t)M * Kp1 * sizeof(bf16));
    bf16*  Wlat_bf = (bf16*)carve((size_t)Hl * Kp1 * sizeof(bf16));
    bf16*  X_bf    = (bf16*)carve((size_t)M * Kp2 * sizeof(bf16));
    bf16*  Wih_bf  = (bf16*)carve((size_t)G * Kp2 * sizeof(bf16));
    bf16*  Whh_bf  = (bf16*)carve((size_t)G * Kp2 * sizeof(bf16));
    float* gates_x = (float*)carve((size_t)M * G * sizeof(float));
    float* H0      = (float*)carve((size_t)L * Hl * sizeof(float));
    float* amu     = (float*)carve((size_t)L * 50 * sizeof(float));
    float* alv     = (float*)carve((size_t)L * 50 * sizeof(float));

    const int T = 256;
    auto nblk = [](long long n, int t) { return (unsigned)((n + t - 1) / t); };

    // 1) convert + pad to bf16
    k_cvt_pad<<<nblk((long long)M * Kp1, T), T, 0, stream>>>(data,  data_bf, M,  D,  Kp1);
    k_cvt_pad<<<nblk((long long)Hl * Kp1, T), T, 0, stream>>>(W_lat, Wlat_bf, Hl, D,  Kp1);
    k_cvt_pad<<<nblk((long long)G * Kp2, T), T, 0, stream>>>(W_ih,  Wih_bf,  G,  Hl, Kp2);
    k_cvt_pad<<<nblk((long long)G * Kp2, T), T, 0, stream>>>(W_hh,  Whh_bf,  G,  Hl, Kp2);
    k_fill_zero_bf16<<<nblk((long long)M * Kp2, T), T, 0, stream>>>(X_bf, (long long)M * Kp2);

    // 2) X = data @ W_lat^T + b_lat   (out bf16, padded ldc=416)
    {
        const int Mg = M / 64, Nt = Hl / 16;           // 64 x 25
        const int waves = Mg * Nt, wpb = 8;
        k_gemm_wmma<true><<<nblk(waves, wpb), wpb * 32, 0, stream>>>(
            data_bf, Kp1, Wlat_bf, Kp1, b_lat, nullptr, X_bf, Kp2, Mg, Nt, Kp1 / 32);
    }

    // 3) gates_x = X @ W_ih^T + b_ih + b_hh   (out f32, ldc=1600)
    {
        const int Mg = M / 64, Nt = G / 16;            // 64 x 100
        const int waves = Mg * Nt, wpb = 8;
        k_gemm_wmma<false><<<nblk(waves, wpb), wpb * 32, 0, stream>>>(
            X_bf, Kp2, Wih_bf, Kp2, b_ih, b_hh, gates_x, G, Mg, Nt, Kp2 / 32);
    }

    // 4) sequential LSTM (single WGP-resident workgroup, WMMA inner GEMM)
    {
        const size_t smem = (size_t)8 * 1600 * 4 + 8 * 400 * 4 + 16 * 416 * 2; // 77312 B
        k_lstm_seq<<<1, 1024, smem, stream>>>(Whh_bf, gates_x, H0, L);
    }

    // 5) latent projections (tiny)
    k_proj<<<nblk(512 * 50, T), T, 0, stream>>>(H0, W_mu, b_mu, W_lv, b_lv, amu, alv);

    // 6) latent scan + KL + outputs
    k_scan<<<1, 64, 0, stream>>>(amu, alv, eps, W_mu, W_lv, out);
}
